// DistillerLSTMCell_67078799229551
// MI455X (gfx1250) — matrix operand order repaired and checked by reading, not compiled, not run on hardware
//
#include <hip/hip_runtime.h>

// ---- plain (trivial) vector types -----------------------------------------
typedef __attribute__((ext_vector_type(16))) __bf16         v16bf;
typedef __attribute__((ext_vector_type(8)))  float          v8f;
typedef __attribute__((ext_vector_type(4)))  float          v4f;
typedef __attribute__((ext_vector_type(4)))  unsigned int   v4u;
typedef __attribute__((ext_vector_type(4)))  unsigned short v4s;

#define B_DIM    4096
#define K_DIM    2048          // IN == H == 2048
#define H_DIM    2048
#define BLOCK_M  256           // 8 waves x 2 sub-tiles x 16 rows
#define KT       32            // WMMA K per step
#define LDS_STR  40            // ushorts per B row: 32 data + 8 pad (80B)
#define BBUF     (4 * 16 * LDS_STR)   // ushorts per B buffer (4 gates x 16 n)

union Frag {                   // 16 bf16 = 32 bytes = two 16B loads
    v16bf bf;
    v4u   u4[2];
};

__device__ __forceinline__ unsigned short f32_to_bf16_rne(float f) {
    unsigned int u = __builtin_bit_cast(unsigned int, f);
    u += 0x7FFFu + ((u >> 16) & 1u);
    return (unsigned short)(u >> 16);
}

__device__ __forceinline__ v4s cvt4(v4f v) {
    v4s s;
    s.x = f32_to_bf16_rne(v.x);
    s.y = f32_to_bf16_rne(v.y);
    s.z = f32_to_bf16_rne(v.z);
    s.w = f32_to_bf16_rne(v.w);
    return s;
}

__device__ __forceinline__ float fast_sigmoid(float x) {
    return 1.0f / (1.0f + __expf(-x));
}

// ---- one-time fp32 -> bf16 conversion pass ---------------------------------
__global__ __launch_bounds__(256) void cvt_f32_bf16(
    const float* __restrict__ src, unsigned short* __restrict__ dst, int n4) {
    int i = blockIdx.x * 256 + threadIdx.x;
    if (i < n4) {
        ((v4s*)dst)[i] = cvt4(((const v4f*)src)[i]);
    }
}

// ---- fused LSTM cell: 4-gate bf16 WMMA GEMM + activations -------------------
__global__ __launch_bounds__(256) void lstm_gemm_wmma(
    const unsigned short* __restrict__ xbf,  const unsigned short* __restrict__ hbf,
    const unsigned short* __restrict__ Wxbf, const unsigned short* __restrict__ Whbf,
    const float* __restrict__ c_prev, const float* __restrict__ b_x,
    const float* __restrict__ b_h,    float* __restrict__ out)
{
    __shared__ unsigned short ldsB[2 * BBUF];   // double-buffered W tile

    const int tid  = threadIdx.x;
    const int wave = tid >> 5;
    const int lane = tid & 31;
    const int half = lane >> 4;
    const int ln   = lane & 15;

    const int m0 = blockIdx.x * BLOCK_M;
    const int n0 = blockIdx.y * 16;

    // async-copy assignment: 256 threads x 16B = full 4KB B tile
    const int cg = tid >> 6;            // gate
    const int cr = (tid >> 2) & 15;     // n-row within gate
    const int cc = tid & 3;             // 16B chunk within 64B row
    const unsigned ldsDst0 =
        (unsigned)(size_t)&ldsB[(cg * 16 + cr) * LDS_STR + cc * 8];
    const unsigned ldsDst1 =
        (unsigned)(size_t)&ldsB[BBUF + (cg * 16 + cr) * LDS_STR + cc * 8];
    const size_t wRowOff = (size_t)(cg * H_DIM + n0 + cr) * K_DIM + cc * 8;

    v8f acc[2][4] = {};                 // [m-subtile][gate]

    #pragma unroll 1
    for (int phase = 0; phase < 2; ++phase) {
        const unsigned short* __restrict__ Abf = phase ? hbf  : xbf;
        const unsigned short* __restrict__ Wbf = phase ? Whbf : Wxbf;

        // preload K panel 0 into buffer 0 (ASYNCcnt -> 1)
        asm volatile("global_load_async_to_lds_b128 %0, %1, off"
                     :: "v"(ldsDst0), "v"(Wbf + wRowOff) : "memory");

        #pragma unroll 1
        for (int k0 = 0; k0 < K_DIM; k0 += KT) {
            const int cur = (k0 >> 5) & 1;
            if (k0 + KT < K_DIM) {
                // issue copy of next K panel into the other buffer, then
                // wait until only that newest copy is outstanding
                const unsigned dstNext = cur ? ldsDst0 : ldsDst1;
                asm volatile("global_load_async_to_lds_b128 %0, %1, off"
                             :: "v"(dstNext), "v"(Wbf + wRowOff + k0 + KT) : "memory");
                asm volatile("s_wait_asynccnt 1" ::: "memory");
            } else {
                asm volatile("s_wait_asynccnt 0" ::: "memory");
            }
            __syncthreads();            // all waves' async chunks landed

            // A fragments, straight from global bf16 (L2-resident)
            Frag fa[2];
            #pragma unroll
            for (int s = 0; s < 2; ++s) {
                const unsigned short* arow =
                    Abf + (size_t)(m0 + wave * 32 + s * 16 + ln) * K_DIM + k0;
                fa[s].u4[0] = *(const v4u*)(arow + half * 8);        // K 0..7 | 8..15
                fa[s].u4[1] = *(const v4u*)(arow + 16 + half * 8);   // K 16..23 | 24..31
            }

            const unsigned short* bbase = cur ? (ldsB + BBUF) : ldsB;
            #pragma unroll
            for (int g = 0; g < 4; ++g) {
                Frag fb;
                const unsigned short* brow = bbase + (g * 16 + ln) * LDS_STR;
                fb.u4[0] = *(const v4u*)(brow + half * 8);           // K 0..7 | 8..15
                fb.u4[1] = *(const v4u*)(brow + 16 + half * 8);      // K 16..23 | 24..31
                #pragma unroll
                for (int s = 0; s < 2; ++s) {
                    acc[s][g] = __builtin_amdgcn_wmma_f32_16x16x32_bf16(
                        false, fa[s].bf, false, fb.bf, (short)0, acc[s][g],
                        false, false);
                }
            }
            __syncthreads();            // everyone done reading buf[cur]
        }
    }

    // ---- epilogue: biases + LSTM elementwise, write h and c ----
    const int col = n0 + ln;
    const float bi = b_x[0 * H_DIM + col] + b_h[0 * H_DIM + col];
    const float bf = b_x[1 * H_DIM + col] + b_h[1 * H_DIM + col];
    const float bg = b_x[2 * H_DIM + col] + b_h[2 * H_DIM + col];
    const float bo = b_x[3 * H_DIM + col] + b_h[3 * H_DIM + col];

    #pragma unroll
    for (int s = 0; s < 2; ++s) {
        #pragma unroll
        for (int r = 0; r < 8; ++r) {
            const int row = m0 + wave * 32 + s * 16 + half * 8 + r;
            const size_t idx = (size_t)row * H_DIM + col;
            const float gi = fast_sigmoid(acc[s][0][r] + bi);
            const float gf = fast_sigmoid(acc[s][1][r] + bf);
            const float gg = tanhf(acc[s][2][r] + bg);
            const float go = fast_sigmoid(acc[s][3][r] + bo);
            const float cp = c_prev[idx];
            const float cv = gf * cp + gi * gg;
            const float hv = go * tanhf(cv);
            out[idx] = hv;
            out[(size_t)B_DIM * H_DIM + idx] = cv;
        }
    }
}

extern "C" void kernel_launch(void* const* d_in, const int* in_sizes, int n_in,
                              void* d_out, int out_size, void* d_ws, size_t ws_size,
                              hipStream_t stream) {
    const float* x      = (const float*)d_in[0];
    const float* h_prev = (const float*)d_in[1];
    const float* c_prev = (const float*)d_in[2];
    const float* W_x    = (const float*)d_in[3];
    const float* b_x    = (const float*)d_in[4];
    const float* W_h    = (const float*)d_in[5];
    const float* b_h    = (const float*)d_in[6];
    float* out = (float*)d_out;

    // bf16 workspace layout (ushort elements)
    unsigned short* ws  = (unsigned short*)d_ws;
    const size_t nXH = (size_t)B_DIM * K_DIM;        // 8M elements each
    const size_t nW  = (size_t)4 * H_DIM * K_DIM;    // 16M elements each
    unsigned short* xbf  = ws;
    unsigned short* hbf  = xbf + nXH;
    unsigned short* Wxbf = hbf + nXH;
    unsigned short* Whbf = Wxbf + nW;

    // one-time fp32 -> bf16 conversion (all counts divisible by 1024)
    cvt_f32_bf16<<<(int)(nXH / 4 / 256), 256, 0, stream>>>(x,      xbf,  (int)(nXH / 4));
    cvt_f32_bf16<<<(int)(nXH / 4 / 256), 256, 0, stream>>>(h_prev, hbf,  (int)(nXH / 4));
    cvt_f32_bf16<<<(int)(nW  / 4 / 256), 256, 0, stream>>>(W_x,    Wxbf, (int)(nW / 4));
    cvt_f32_bf16<<<(int)(nW  / 4 / 256), 256, 0, stream>>>(W_h,    Whbf, (int)(nW / 4));

    dim3 grid(B_DIM / BLOCK_M, H_DIM / 16);   // 16 x 128 workgroups
    lstm_gemm_wmma<<<grid, 256, 0, stream>>>(xbf, hbf, Wxbf, Whbf,
                                             c_prev, b_x, b_h, out);
}
